// SNNResidualBlock_34265249087670
// MI455X (gfx1250) — matrix operand order repaired and checked by reading, not compiled, not run on hardware
//
#include <hip/hip_runtime.h>

typedef __attribute__((ext_vector_type(16))) __bf16 v16bf;
typedef __attribute__((ext_vector_type(8)))  __bf16 v8bf;
typedef __attribute__((ext_vector_type(8)))  float  v8f;

#define BATCH 32
#define COUT  256
#define LEN   4096
#define LPAD  4098         // padded rows per (b, ci-chunk): guard row at each end
#define MT    64           // Cout tile per block
#define NT    128          // L tile per block
#define KC    32           // input-channel chunk (WMMA K depth)
#define LSTR  40           // LDS row stride in halves (80B: 16B-aligned, 20-bank stride)

// ---------------------------------------------------------------------------
// CDNA5 async global->LDS copies (ASYNCcnt path). 16B per lane; the LDS-side
// per-lane scatter lets us insert the 80B row padding during the copy itself.
// ---------------------------------------------------------------------------
__device__ __forceinline__ void async_b128(uint32_t lds_off, const __bf16* g)
{
    asm volatile("global_load_async_to_lds_b128 %0, %1, off"
                 :: "v"(lds_off), "v"(g) : "memory");
}
template<int N>
__device__ __forceinline__ void wait_async()
{
    asm volatile("s_wait_asynccnt %0" :: "i"(N) : "memory");
}
__device__ __forceinline__ uint32_t lds_offset(const void* p)
{
    return (uint32_t)(uintptr_t)p;   // low 32 bits of flat LDS aperture = LDS offset
}

// ---------------------------------------------------------------------------
// Pack x (f32 [B][Cin][L]) -> bf16 xpack [b][ci/32][LPAD][kq], LDS-transposed
// so conv staging is one contiguous async copy per chunk.
// ---------------------------------------------------------------------------
__global__ __launch_bounds__(256)
void pack_x_kernel(const float* __restrict__ x, __bf16* __restrict__ xp, int nch)
{
    __shared__ __align__(16) __bf16 tile[32 * 64];
    const int tid = threadIdx.x;
    const int l0  = blockIdx.x * 64;
    const int c   = blockIdx.y;
    const int b   = blockIdx.z;
    {   // coalesced f32 reads, convert, LDS tile [ci][l]
        const int ci = tid >> 3;
        const int ls = (tid & 7) * 8;
        const float* src = x + ((size_t)b * (nch * 32) + (c * 32 + ci)) * (size_t)LEN + l0 + ls;
        v8bf v;
#pragma unroll
        for (int k = 0; k < 8; ++k) v[k] = (__bf16)src[k];
        *(v8bf*)&tile[ci * 64 + ls] = v;
    }
    __syncthreads();
    {   // coalesced 16B packed writes: row l, kq contiguous
        const int lr  = tid >> 2;
        const int seg = (tid & 3) * 8;
        v8bf v;
#pragma unroll
        for (int k = 0; k < 8; ++k) v[k] = tile[(seg + k) * 64 + lr];
        const size_t off = (((size_t)(b * nch + c)) * LPAD + (l0 + lr + 1)) * KC + seg;
        *(v8bf*)&xp[off] = v;
    }
}

__global__ void zero_guards_kernel(__bf16* base, int nbc)
{
    const int idx = blockIdx.x * 256 + threadIdx.x;
    if (idx >= nbc * 64) return;
    const int bc   = idx >> 6;
    const int rest = idx & 63;
    const size_t row = (rest & 32) ? (LPAD - 1) : 0;
    base[((size_t)bc * LPAD + row) * KC + (rest & 31)] = (__bf16)0.0f;
}

// Pack weights f32 [Cout][Cin][KW] -> bf16 [mb][chunk][kk][m][kq] (contiguous
// per (mb,chunk) block => one straight async copy in the conv kernel).
__global__ void pack_w_kernel(const float* __restrict__ w, __bf16* __restrict__ wpk,
                              int cin, int kw)
{
    const int nch   = cin / KC;
    const int idx   = blockIdx.x * 256 + threadIdx.x;
    const int total = COUT * cin * kw;
    if (idx >= total) return;
    const int kq = idx & 31;
    int r = idx >> 5;
    const int m  = r % MT;  r /= MT;
    const int kk = r % kw;  r /= kw;
    const int c  = r % nch;
    const int mb = r / nch;
    wpk[idx] = (__bf16)w[(((size_t)(mb * MT + m)) * cin + (c * KC + kq)) * kw + kk];
}

// ---------------------------------------------------------------------------
// Implicit-GEMM conv1d via v_wmma_f32_16x16x32_bf16 with double-buffered
// async-LDS staging: copies for chunk c+1 fly under the WMMA loop of chunk c
// (s_wait_asynccnt <n> keeps them in flight).
// Bias dropped: exactly cancelled by training-mode BN mean subtraction.
// ---------------------------------------------------------------------------
template<int CIN, int KW, bool OUTPACKED>
__global__ __launch_bounds__(256)
void conv_wmma_async(const __bf16* __restrict__ xp, const __bf16* __restrict__ wp,
                     __bf16* __restrict__ yout, __bf16* __restrict__ ypack,
                     float* __restrict__ stat_sum, float* __restrict__ stat_sq)
{
    constexpr int NCH   = CIN / KC;
    constexpr int PAD   = (KW - 1) / 2;
    constexpr int LLD   = NT + KW - 1;       // haloed columns
    constexpr int WROWS = KW * MT;
    constexpr int WSEGS = WROWS * 4;         // 16B segments (row = 64B data)
    constexpr int XSEGS = LLD * 4;
    constexpr int WIT   = (WSEGS + 255) / 256;
    constexpr int XIT   = (XSEGS + 255) / 256;
    constexpr int NASY  = WIT + XIT;         // async ops per wave per chunk

    __shared__ __align__(16) __bf16 sW[2][WROWS * LSTR];
    __shared__ __align__(16) __bf16 sX[2][LLD * LSTR];

    const int tid  = threadIdx.x;
    const int lane = tid & 31;
    const int wave = tid >> 5;
    const int wm   = (wave >> 2) * 32;       // 2 waves along M
    const int wn   = (wave & 3) * 32;        // 4 waves along N

    const int b  = blockIdx.z;
    const int mb = blockIdx.y;
    const int m0 = mb * MT;
    const int l0 = blockIdx.x * NT;

    v8f acc[2][2];
#pragma unroll
    for (int i = 0; i < 2; ++i)
#pragma unroll
        for (int j = 0; j < 2; ++j)
            acc[i][j] = v8f{};

    auto preload = [&](int c, int buf) {
        const size_t wbase = (size_t)(mb * NCH + c) * ((size_t)WROWS * KC);
        const uint32_t sWo = lds_offset(&sW[buf][0]);
#pragma unroll
        for (int it = 0; it < WIT; ++it) {
            int idx = it * 256 + tid;
            if (idx >= WSEGS) idx -= WSEGS;  // keep all lanes active (dup copy ok)
            async_b128(sWo + (uint32_t)(idx >> 2) * (LSTR * 2) + (uint32_t)(idx & 3) * 16,
                       wp + wbase + (size_t)idx * 8);
        }
        const size_t xbase = ((size_t)(b * NCH + c) * LPAD + (size_t)(l0 - PAD + 1)) * KC;
        const uint32_t sXo = lds_offset(&sX[buf][0]);
#pragma unroll
        for (int it = 0; it < XIT; ++it) {
            int idx = it * 256 + tid;
            if (idx >= XSEGS) idx -= XSEGS;
            async_b128(sXo + (uint32_t)(idx >> 2) * (LSTR * 2) + (uint32_t)(idx & 3) * 16,
                       xp + xbase + (size_t)idx * 8);
        }
    };

    preload(0, 0);
    for (int c = 0; c < NCH; ++c) {
        const int cur = c & 1;
        if (c + 1 < NCH) {
            preload(c + 1, cur ^ 1);     // in flight during this chunk's WMMAs
            wait_async<NASY>();          // chunk-c copies landed; c+1 still flying
        } else {
            wait_async<0>();
        }
        __syncthreads();

        const __bf16* sWc = &sW[cur][0];
        const __bf16* sXc = &sX[cur][0];
#pragma unroll
        for (int kk = 0; kk < KW; ++kk) {
            // A fragment (16x32 bf16): lanes 0-15 K=0..7/16..23, lanes 16-31 +8
            const int am = lane & 15;
            const int kb = (lane >> 4) * 8;
            const v8bf* wr0 = (const v8bf*)&sWc[(kk * MT + wm + am) * LSTR + kb];
            const v8bf* wr1 = (const v8bf*)&sWc[(kk * MT + wm + 16 + am) * LSTR + kb];
            const v16bf a0 = __builtin_shufflevector(wr0[0], wr0[2],
                0,1,2,3,4,5,6,7,8,9,10,11,12,13,14,15);
            const v16bf a1 = __builtin_shufflevector(wr1[0], wr1[2],
                0,1,2,3,4,5,6,7,8,9,10,11,12,13,14,15);
            // B fragment (32x16 bf16): lanes 0-15 K=0..7/16..23, lanes 16-31 +8
            const int ln  = lane & 15;
            const int kb2 = (lane >> 4) * 8;
            const v8bf* xr0 = (const v8bf*)&sXc[(wn + ln + kk) * LSTR + kb2];
            const v8bf* xr1 = (const v8bf*)&sXc[(wn + 16 + ln + kk) * LSTR + kb2];
            const v16bf b0 = __builtin_shufflevector(xr0[0], xr0[2],
                0,1,2,3,4,5,6,7,8,9,10,11,12,13,14,15);
            const v16bf b1 = __builtin_shufflevector(xr1[0], xr1[2],
                0,1,2,3,4,5,6,7,8,9,10,11,12,13,14,15);

            acc[0][0] = __builtin_amdgcn_wmma_f32_16x16x32_bf16(false, a0, false, b0, (short)0, acc[0][0], false, false);
            acc[0][1] = __builtin_amdgcn_wmma_f32_16x16x32_bf16(false, a0, false, b1, (short)0, acc[0][1], false, false);
            acc[1][0] = __builtin_amdgcn_wmma_f32_16x16x32_bf16(false, a1, false, b0, (short)0, acc[1][0], false, false);
            acc[1][1] = __builtin_amdgcn_wmma_f32_16x16x32_bf16(false, a1, false, b1, (short)0, acc[1][1], false, false);
        }
        __syncthreads();   // all waves done reading buf[cur] before it is refilled
    }

    // ---- epilogue: C/D layout VGPR r -> M = r + 8*(lane>=16), N = lane&15
#pragma unroll
    for (int i = 0; i < 2; ++i) {
        const int co_base = m0 + wm + 16 * i + 8 * (lane >> 4);
        float s_sum[8] = {0,0,0,0,0,0,0,0};
        float s_sq [8] = {0,0,0,0,0,0,0,0};
#pragma unroll
        for (int j = 0; j < 2; ++j) {
            const int l = l0 + wn + 16 * j + (lane & 15);
#pragma unroll
            for (int r = 0; r < 8; ++r) {
                const float v = acc[i][j][r];
                const int co  = co_base + r;
                if constexpr (OUTPACKED) {
                    const size_t off =
                        (((size_t)(b * (COUT / KC) + (co >> 5))) * LPAD + (l + 1)) * KC + (co & 31);
                    ypack[off] = (__bf16)v;
                } else {
                    yout[((size_t)b * COUT + co) * (size_t)LEN + l] = (__bf16)v;
                }
                s_sum[r] += v;
                s_sq [r] += v * v;
            }
        }
#pragma unroll
        for (int r = 0; r < 8; ++r) {
            atomicAdd(&stat_sum[co_base + r], s_sum[r]);
            atomicAdd(&stat_sq [co_base + r], s_sq [r]);
        }
    }
}

// ---------------------------------------------------------------------------
__global__ void zero_stats_kernel(float* st, int n)
{
    const int i = blockIdx.x * blockDim.x + threadIdx.x;
    if (i < n) st[i] = 0.0f;
}

__global__ void bn_finalize_kernel(const float* __restrict__ sum, const float* __restrict__ sq,
                                   const float* __restrict__ gamma, const float* __restrict__ beta,
                                   float* __restrict__ scale, float* __restrict__ shift)
{
    const int c = blockIdx.x * blockDim.x + threadIdx.x;
    if (c >= COUT) return;
    const float n    = (float)BATCH * (float)LEN;
    const float mean = sum[c] / n;
    const float var  = sq[c] / n - mean * mean;
    const float s    = gamma[c] * rsqrtf(var + 1e-5f);
    scale[c] = s;
    shift[c] = beta[c] - mean * s;
}

__device__ __forceinline__ float lif_mean(float u, int T)
{
    float v = 0.0f, s = 0.0f;
    for (int t = 0; t < T; ++t) {
        v = 0.5f * (v + u);                      // tau = 2
        const float spk = (v >= 1.0f) ? 1.0f : 0.0f;
        s += spk;
        v *= (1.0f - spk);                       // hard reset
    }
    return s / (float)T;
}

// BN + LIF in packed space (in-place over xpack2); also zeroes the guard rows
// so conv2 sees correct zero padding.
__global__ __launch_bounds__(256)
void bn_lif_packed_kernel(__bf16* __restrict__ xp2,
                          const float* __restrict__ scale, const float* __restrict__ shift,
                          const int* __restrict__ Tp)
{
    const size_t i8  = ((size_t)blockIdx.x * 256 + threadIdx.x) * 8;
    const int    kq0 = (int)(i8 & 31);
    const size_t row = (i8 >> 5) % LPAD;
    const int  chunk = (int)((i8 >> 5) / LPAD) & 7;
    if (row == 0 || row == LPAD - 1) { *(v8bf*)&xp2[i8] = v8bf{}; return; }
    const int T   = Tp[0];
    const int co0 = chunk * 32 + kq0;
    const v8bf v = *(const v8bf*)&xp2[i8];
    v8bf o;
#pragma unroll
    for (int k = 0; k < 8; ++k) {
        const float u = (float)v[k] * scale[co0 + k] + shift[co0 + k];
        o[k] = (__bf16)lif_mean(u, T);
    }
    *(v8bf*)&xp2[i8] = o;
}

// out = LIF(BN2(conv2)) + BNd(dsconv), fully coalesced f32 output
__global__ __launch_bounds__(256)
void final_kernel(const __bf16* __restrict__ y2, const __bf16* __restrict__ yd,
                  const float* __restrict__ sc2, const float* __restrict__ sh2,
                  const float* __restrict__ scd, const float* __restrict__ shd,
                  const int* __restrict__ Tp, float* __restrict__ out)
{
    const size_t i8 = ((size_t)blockIdx.x * 256 + threadIdx.x) * 8;
    const int co = (int)((i8 >> 12) & (COUT - 1));
    const int T  = Tp[0];
    const float s2 = sc2[co], h2 = sh2[co];
    const float sd = scd[co], hd = shd[co];
    const v8bf a = *(const v8bf*)&y2[i8];
    const v8bf r = *(const v8bf*)&yd[i8];
#pragma unroll
    for (int k = 0; k < 8; ++k) {
        const float u   = (float)a[k] * s2 + h2;
        const float res = (float)r[k] * sd + hd;
        out[i8 + k] = lif_mean(u, T) + res;
    }
}

// ---------------------------------------------------------------------------
extern "C" void kernel_launch(void* const* d_in, const int* in_sizes, int n_in,
                              void* d_out, int out_size, void* d_ws, size_t ws_size,
                              hipStream_t stream)
{
    const float* x  = (const float*)d_in[0];
    const float* w1 = (const float*)d_in[1];
    const float* g1 = (const float*)d_in[3];
    const float* b1 = (const float*)d_in[4];
    const float* w2 = (const float*)d_in[5];
    const float* g2 = (const float*)d_in[7];
    const float* b2 = (const float*)d_in[8];
    const float* wd = (const float*)d_in[9];
    const float* gd = (const float*)d_in[11];
    const float* bd = (const float*)d_in[12];
    const int*   T  = (const int*)d_in[13];

    const size_t actElems = (size_t)BATCH * COUT * LEN;       // 33,554,432
    const size_t X1H = (size_t)BATCH * 4 * LPAD * KC;         // xpack1 halves
    const size_t X2H = (size_t)BATCH * 8 * LPAD * KC;         // xpack2 halves
    const size_t W1H = (size_t)COUT * 128 * 3;
    const size_t W2H = (size_t)COUT * 256 * 3;
    const size_t WDH = (size_t)COUT * 128 * 1;

    char* ws = (char*)d_ws;
    size_t off = 0;
    auto take = [&](size_t bytes) { size_t o = off; off += (bytes + 255) & ~(size_t)255; return o; };
    __bf16* xp1 = (__bf16*)(ws + take(X1H * 2));
    __bf16* xp2 = (__bf16*)(ws + take(X2H * 2));
    __bf16* wp1 = (__bf16*)(ws + take(W1H * 2));
    __bf16* wp2 = (__bf16*)(ws + take(W2H * 2));
    __bf16* wpd = (__bf16*)(ws + take(WDH * 2));
    __bf16* y2  = (__bf16*)(ws + take(actElems * 2));
    __bf16* yd  = (__bf16*)(ws + take(actElems * 2));
    float*  st  = (float*)(ws + take(3072 * 4));
    float *sum1 = st,        *sq1 = st + 256;
    float *sum2 = st + 512,  *sq2 = st + 768;
    float *sumd = st + 1024, *sqd = st + 1280;
    float *sc1  = st + 1536, *sh1 = st + 1792;
    float *sc2  = st + 2048, *sh2 = st + 2304;
    float *scd  = st + 2560, *shd = st + 2816;

    // prep: stats, guards, packed inputs/weights
    zero_stats_kernel<<<6, 256, 0, stream>>>(st, 1536);
    zero_guards_kernel<<<(BATCH * 4 * 64 + 255) / 256, 256, 0, stream>>>(xp1, BATCH * 4);
    pack_x_kernel<<<dim3(LEN / 64, 4, BATCH), 256, 0, stream>>>(x, xp1, 4);
    pack_w_kernel<<<(int)(W1H / 256), 256, 0, stream>>>(w1, wp1, 128, 3);
    pack_w_kernel<<<(int)(W2H / 256), 256, 0, stream>>>(w2, wp2, 256, 3);
    pack_w_kernel<<<(int)(WDH / 256), 256, 0, stream>>>(wd, wpd, 128, 1);

    const dim3 cgrid(LEN / NT, COUT / MT, BATCH);             // 32 x 4 x 32
    const int  ewBlocks2 = (int)(X2H / 8 / 256);              // bn_lif (packed)
    const int  ewBlocksO = (int)(actElems / 8 / 256);         // final

    // layer 1 (output written straight into packed conv2-input layout)
    conv_wmma_async<128, 3, true><<<cgrid, 256, 0, stream>>>(xp1, wp1, nullptr, xp2, sum1, sq1);
    bn_finalize_kernel<<<1, 256, 0, stream>>>(sum1, sq1, g1, b1, sc1, sh1);
    bn_lif_packed_kernel<<<ewBlocks2, 256, 0, stream>>>(xp2, sc1, sh1, T);

    // layer 2 + downsample
    conv_wmma_async<256, 3, false><<<cgrid, 256, 0, stream>>>(xp2, wp2, y2, nullptr, sum2, sq2);
    conv_wmma_async<128, 1, false><<<cgrid, 256, 0, stream>>>(xp1, wpd, yd, nullptr, sumd, sqd);
    bn_finalize_kernel<<<1, 256, 0, stream>>>(sum2, sq2, g2, b2, sc2, sh2);
    bn_finalize_kernel<<<1, 256, 0, stream>>>(sumd, sqd, gd, bd, scd, shd);

    final_kernel<<<ewBlocksO, 256, 0, stream>>>(y2, yd, sc2, sh2, scd, shd, T, (float*)d_out);
}
